// MultiHeadAttention_24223615550138
// MI455X (gfx1250) — matrix operand order repaired and checked by reading, not compiled, not run on hardware
//
#include <hip/hip_runtime.h>

#define D_HEAD  64
#define D_MODEL 1024
#define N_HEADS 16
#define BATCH   4
#define SEQ     2048
#define BS      (BATCH * SEQ)   // 8192

typedef __bf16 bf16_t;
typedef __attribute__((ext_vector_type(16))) __bf16 v16bf;
typedef __attribute__((ext_vector_type(8)))  float  v8f;

union FragBF16 { v16bf v; bf16_t e[16]; };
union FragF32  { v8f  v; float  e[8];  };

#define WMMA_BF16(A, B, C) \
  __builtin_amdgcn_wmma_f32_16x16x32_bf16(false, (A), false, (B), (short)0, (C), false, false)

// ---------------------------------------------------------------------------
// Swizzled-fragment index helpers (WMMA 16x16x32 bf16, wave32; this code's contiguous k order, k = 16*(lane/16) + elem).
// A swizzle: lane = (row&15) + 16*((k>>4)&1) ; elem = k&15
// B swizzle: lane = (n&15)   + 16*((k>>4)&1) ; elem = k&15
// Each lane's 16 elements are contiguous (32B) -> two b128 loads.
// ---------------------------------------------------------------------------
__device__ __forceinline__ size_t aswz(int kblocks, int row, int k) {
  const int kl   = k & 31;
  const int lane = (row & 15) + (((kl >> 4) & 1) << 4);
  const int e    = kl & 15;
  return (((size_t)(row >> 4) * kblocks + (k >> 5)) * 32 + lane) * 16 + e;
}

__device__ __forceinline__ v16bf frag_ld16(const bf16_t* p) {
  FragBF16 f;
  *(uint4*)&f.e[0] = *(const uint4*)p;
  *(uint4*)&f.e[8] = *(const uint4*)(p + 8);
  return f.v;
}

__device__ __forceinline__ v16bf frag_a_gsw(const bf16_t* base, int kblocks,
                                            int rb, int kb, int lane) {
  return frag_ld16(base + (((size_t)rb * kblocks + kb) * 32 + lane) * 16);
}

// ---------------------------------------------------------------------------
// CDNA5 async global->LDS copy (ASYNCcnt path), per-lane 16B.
// ---------------------------------------------------------------------------
__device__ __forceinline__ void async_copy_b128(const bf16_t* gsrc, bf16_t* ldst) {
  const unsigned lds_off = (unsigned)(uintptr_t)ldst;   // low 32 bits = LDS offset
  asm volatile("global_load_async_to_lds_b128 %0, %1, off"
               :: "v"(lds_off), "v"(gsrc)
               : "memory");
}
__device__ __forceinline__ void wait_async0() {
  asm volatile("s_wait_asynccnt 0" ::: "memory");
}
__device__ __forceinline__ void wait_async8() {   // keep newest 8 in flight
  asm volatile("s_wait_asynccnt 8" ::: "memory");
}

// ---------------------------------------------------------------------------
// fp32 -> bf16 conversion with A-fragment swizzle. One thread per 8-k chunk.
// ---------------------------------------------------------------------------
__global__ void k_cvt_swA(const float* __restrict__ in, bf16_t* __restrict__ out,
                          int R, int K) {
  const int chunks = K >> 3;
  const int total  = R * chunks;
  const int stride = gridDim.x * blockDim.x;
  for (int t = blockIdx.x * blockDim.x + threadIdx.x; t < total; t += stride) {
    const int row = t / chunks;
    const int k   = (t - row * chunks) * 8;
    const int kl  = k & 31;
    const int lane = (row & 15) + (((kl >> 4) & 1) << 4);
    const int e0   = kl & 15;          // 0 or 8
    const float* s = in + (size_t)row * K + k;
    __align__(16) bf16_t v8[8];
#pragma unroll
    for (int j = 0; j < 8; ++j) v8[j] = (bf16_t)s[j];
    bf16_t* d = out + (((size_t)(row >> 4) * (K >> 5) + (k >> 5)) * 32 + lane) * 16 + e0;
    *(uint4*)d = *(const uint4*)v8;
  }
}

// fp32 -> bf16 with B-fragment swizzle. Input [G][K][N] row-major.
__global__ void k_cvt_swB(const float* __restrict__ in, bf16_t* __restrict__ out,
                          int G, int K, int N) {
  const int perG   = (K >> 3) * N;
  const int total  = G * perG;
  const int stride = gridDim.x * blockDim.x;
  for (int t = blockIdx.x * blockDim.x + threadIdx.x; t < total; t += stride) {
    const int g  = t / perG;
    const int u  = t - g * perG;
    const int kc = u / N;
    const int n  = u - kc * N;
    const int k  = kc * 8;
    const int lane = (n & 15) + (((k >> 4) & 1) << 4);
    const int e0   = k & 15;           // 0 or 8
    __align__(16) bf16_t v8[8];
#pragma unroll
    for (int j = 0; j < 8; ++j)
      v8[j] = (bf16_t)in[((size_t)g * K + k + j) * N + n];
    bf16_t* d = out + (size_t)g * K * N +
                (((size_t)(n >> 4) * (K >> 5) + (k >> 5)) * 32 + lane) * 16 + e0;
    *(uint4*)d = *(const uint4*)v8;
  }
}

// ---------------------------------------------------------------------------
// GEMM core: 32x64 tile per wave, unroll-by-2 register double buffering,
// FULLY UNROLLED so every load is base-pointer + immediate offset
// (k-step stride = 512 elements = 1024 B; max offset 31 KB << 24-bit IOFFSET).
// ---------------------------------------------------------------------------
template <int KB>
__device__ __forceinline__ void gemm_tile_32x64(
    const bf16_t* __restrict__ Abase, const bf16_t* __restrict__ Bbase,
    int rb0, int rb1, int nb0, int lane, FragF32 acc[8]) {
  constexpr int KS = 32 * 16;   // elements per k-step per fragment block
  // Hoisted per-operand base pointers (lane offset folded in once).
  const bf16_t* pA0 = Abase + (size_t)rb0 * (KB * KS) + lane * 16;
  const bf16_t* pA1 = Abase + (size_t)rb1 * (KB * KS) + lane * 16;
  const bf16_t* pB0 = Bbase + (size_t)(nb0 + 0) * (KB * KS) + lane * 16;
  const bf16_t* pB1 = Bbase + (size_t)(nb0 + 1) * (KB * KS) + lane * 16;
  const bf16_t* pB2 = Bbase + (size_t)(nb0 + 2) * (KB * KS) + lane * 16;
  const bf16_t* pB3 = Bbase + (size_t)(nb0 + 3) * (KB * KS) + lane * 16;

  v16bf a0, a1, b0, b1, b2, b3;   // even set
  v16bf c0, c1, d0, d1, d2, d3;   // odd set

  a0 = frag_ld16(pA0);
  a1 = frag_ld16(pA1);
  b0 = frag_ld16(pB0);
  b1 = frag_ld16(pB1);
  b2 = frag_ld16(pB2);
  b3 = frag_ld16(pB3);

#pragma unroll
  for (int kb = 0; kb < KB; kb += 2) {
    // prefetch odd k-step (immediate offsets)
    c0 = frag_ld16(pA0 + (kb + 1) * KS);
    c1 = frag_ld16(pA1 + (kb + 1) * KS);
    d0 = frag_ld16(pB0 + (kb + 1) * KS);
    d1 = frag_ld16(pB1 + (kb + 1) * KS);
    d2 = frag_ld16(pB2 + (kb + 1) * KS);
    d3 = frag_ld16(pB3 + (kb + 1) * KS);
    // compute even k-step
    acc[0].v = WMMA_BF16(a0, b0, acc[0].v);
    acc[1].v = WMMA_BF16(a0, b1, acc[1].v);
    acc[2].v = WMMA_BF16(a0, b2, acc[2].v);
    acc[3].v = WMMA_BF16(a0, b3, acc[3].v);
    acc[4].v = WMMA_BF16(a1, b0, acc[4].v);
    acc[5].v = WMMA_BF16(a1, b1, acc[5].v);
    acc[6].v = WMMA_BF16(a1, b2, acc[6].v);
    acc[7].v = WMMA_BF16(a1, b3, acc[7].v);
    // prefetch next even k-step (wrap keeps it branchless & in-bounds)
    const int kn = (kb + 2) & (KB - 1);
    a0 = frag_ld16(pA0 + kn * KS);
    a1 = frag_ld16(pA1 + kn * KS);
    b0 = frag_ld16(pB0 + kn * KS);
    b1 = frag_ld16(pB1 + kn * KS);
    b2 = frag_ld16(pB2 + kn * KS);
    b3 = frag_ld16(pB3 + kn * KS);
    // compute odd k-step
    acc[0].v = WMMA_BF16(c0, d0, acc[0].v);
    acc[1].v = WMMA_BF16(c0, d1, acc[1].v);
    acc[2].v = WMMA_BF16(c0, d2, acc[2].v);
    acc[3].v = WMMA_BF16(c0, d3, acc[3].v);
    acc[4].v = WMMA_BF16(c1, d0, acc[4].v);
    acc[5].v = WMMA_BF16(c1, d1, acc[5].v);
    acc[6].v = WMMA_BF16(c1, d2, acc[6].v);
    acc[7].v = WMMA_BF16(c1, d3, acc[7].v);
  }
}

// ---------------------------------------------------------------------------
// QKV projection, LDS-free, software-pipelined.
// Q stored A-swizzled per head; K stored [s][d]; V stored transposed [d][s].
// ---------------------------------------------------------------------------
__global__ __launch_bounds__(128) void k_qkv(
    const bf16_t* __restrict__ Xsw,
    const bf16_t* __restrict__ Wq, const bf16_t* __restrict__ Wk, const bf16_t* __restrict__ Wv,
    bf16_t* __restrict__ Q, bf16_t* __restrict__ K, bf16_t* __restrict__ V) {
  const int rowBlk = blockIdx.x;        // 128 rows of X per block
  const int h      = blockIdx.y;
  const int which  = blockIdx.z;        // 0=Q 1=K 2=V
  const bf16_t* W = (which == 0) ? Wq : (which == 1) ? Wk : Wv;
  bf16_t*     Dst = (which == 0) ? Q  : (which == 1) ? K  : V;
  const bf16_t* Wh = W + (size_t)h * D_MODEL * D_HEAD;

  const int tid  = threadIdx.x;
  const int wave = tid >> 5;
  const int lane = tid & 31;
  const int rb0  = rowBlk * 8 + wave * 2;

  FragF32 acc[8] = {};
  gemm_tile_32x64<D_MODEL / 32>(Xsw, Wh, rb0, rb0 + 1, 0, lane, acc);

  const int colN = lane & 15;
  const int mb   = (lane >> 4) * 8;
#pragma unroll
  for (int half = 0; half < 2; ++half)
#pragma unroll
    for (int n = 0; n < 4; ++n)
#pragma unroll
      for (int r = 0; r < 8; ++r) {
        const int row = rowBlk * 128 + wave * 32 + half * 16 + mb + r;  // b*SEQ + s
        const int b   = row >> 11;
        const int s   = row & (SEQ - 1);
        const int d   = n * 16 + colN;
        const size_t headOff = (((size_t)b * N_HEADS + h) * SEQ) * D_HEAD;
        size_t idx;
        if (which == 0)      idx = headOff + aswz(D_HEAD / 32, s, d);    // Q A-swizzled
        else if (which == 1) idx = headOff + (size_t)s * D_HEAD + d;     // K [s][d]
        else                 idx = headOff + (size_t)d * SEQ + s;        // V [d][s]
        Dst[idx] = (bf16_t)acc[half * 4 + n].e[r];
      }
}

// ---------------------------------------------------------------------------
// Causal flash attention, 64 q rows per block, 4 waves.
// K/V tiles double-buffered in LDS via CDNA5 async-to-LDS (ASYNCcnt overlap).
// ---------------------------------------------------------------------------
__global__ __launch_bounds__(128) void k_attn(
    const bf16_t* __restrict__ Q, const bf16_t* __restrict__ K,
    const bf16_t* __restrict__ V, bf16_t* __restrict__ Csw) {
  const int qBlk = blockIdx.x;
  const int h    = blockIdx.y;
  const int b    = blockIdx.z;
  const int tid  = threadIdx.x;
  const int wave = tid >> 5;
  const int lane = tid & 31;
  const int qbase = qBlk * 64;

  const size_t headOff = (((size_t)b * N_HEADS + h) * SEQ) * D_HEAD;
  const bf16_t* Qsw = Q + headOff;   // A-swizzled
  const bf16_t* Kh  = K + headOff;   // [s][d]
  const bf16_t* VT  = V + headOff;   // [d][s]

  __shared__ __align__(16) bf16_t Kt[2][64 * 64];   // [kv][d]
  __shared__ __align__(16) bf16_t Vt[2][64 * 64];   // [d][kv]
  __shared__ __align__(16) bf16_t Pt[4][16 * 64];

  const int sr = tid >> 1;             // staging row 0..63
  const int scol = (tid & 1) * 32;     // staging col 0 / 32
  auto stage = [&](int kv0, int buf) {
    const bf16_t* ks = Kh + (size_t)(kv0 + sr) * D_HEAD + scol;
    const bf16_t* vs = VT + (size_t)sr * SEQ + kv0 + scol;
    bf16_t* kd = &Kt[buf][sr * 64 + scol];
    bf16_t* vd = &Vt[buf][sr * 64 + scol];
#pragma unroll
    for (int u = 0; u < 4; ++u) {
      async_copy_b128(ks + u * 8, kd + u * 8);   // 8 async b128 per thread total
      async_copy_b128(vs + u * 8, vd + u * 8);
    }
  };

  // Q fragments (A-layout) straight from swizzled global
  FragBF16 qa[2];
  {
    const int rb = qBlk * 4 + wave;
    qa[0].v = frag_a_gsw(Qsw, 2, rb, 0, lane);
    qa[1].v = frag_a_gsw(Qsw, 2, rb, 1, lane);
  }

  FragF32 o[4] = {};
  float m_run[8], l_run[8];
#pragma unroll
  for (int r = 0; r < 8; ++r) { m_run[r] = -1e30f; l_run[r] = 0.0f; }

  const int colN = lane & 15;
  const int mb   = (lane >> 4) * 8;
  const int kbB  = (lane >> 4) << 4;
  const int kbA  = (lane >> 4) << 4;

  stage(0, 0);                         // prologue: tile 0 in flight

  for (int t = 0; t <= qBlk; ++t) {    // causal: kv tiles 0..qBlk
    const int kv0 = t * 64;
    if (t < qBlk) {                    // overlap: issue tile t+1 into other buffer
      stage(kv0 + 64, (t + 1) & 1);
      wait_async8();                   // in-order: tile t's 8 copies have landed
    } else {
      wait_async0();
    }
    __syncthreads();                   // tile t visible to all waves
    const bf16_t* Kc = Kt[t & 1];
    const bf16_t* Vc = Vt[t & 1];

    // S = Q.K^T
    FragF32 sc[4] = {};
#pragma unroll
    for (int dslc = 0; dslc < 2; ++dslc)
#pragma unroll
      for (int n = 0; n < 4; ++n) {
        const v16bf kf = frag_ld16(&Kc[(n * 16 + colN) * 64 + dslc * 32 + kbB]);
        sc[n].v = WMMA_BF16(qa[dslc].v, kf, sc[n].v);
      }

    // online softmax
    float p[4][8];
    float alpha[8];
#pragma unroll
    for (int r = 0; r < 8; ++r) {
      const int qrow = qbase + wave * 16 + mb + r;
      float v4[4];
      float mx = -1e30f;
#pragma unroll
      for (int n = 0; n < 4; ++n) {
        const int kv = kv0 + n * 16 + colN;
        float s = sc[n].e[r] * 0.125f;
        if (kv > qrow) s = -1e30f;
        v4[n] = s;
        mx = fmaxf(mx, s);
      }
#pragma unroll
      for (int off = 1; off < 16; off <<= 1)
        mx = fmaxf(mx, __shfl_xor(mx, off, 32));
      const float mnew = fmaxf(m_run[r], mx);
      alpha[r] = __expf(m_run[r] - mnew);
      float rsum = 0.0f;
#pragma unroll
      for (int n = 0; n < 4; ++n) {
        const float pe = __expf(v4[n] - mnew);
        p[n][r] = pe;
        rsum += pe;
      }
#pragma unroll
      for (int off = 1; off < 16; off <<= 1)
        rsum += __shfl_xor(rsum, off, 32);
      l_run[r] = l_run[r] * alpha[r] + rsum;
      m_run[r] = mnew;
    }

#pragma unroll
    for (int n = 0; n < 4; ++n)
#pragma unroll
      for (int r = 0; r < 8; ++r) {
        o[n].e[r] *= alpha[r];
        Pt[wave][(mb + r) * 64 + n * 16 + colN] = (bf16_t)p[n][r];
      }

    // O += P.V
#pragma unroll
    for (int ks2 = 0; ks2 < 2; ++ks2) {
      FragBF16 pf;
      {
        const bf16_t* pp = &Pt[wave][(lane & 15) * 64 + ks2 * 32 + kbA];
        *(uint4*)&pf.e[0] = *(const uint4*)pp;
        *(uint4*)&pf.e[8] = *(const uint4*)(pp + 8);
      }
#pragma unroll
      for (int n = 0; n < 4; ++n) {
        const v16bf vf = frag_ld16(&Vc[(n * 16 + colN) * 64 + ks2 * 32 + kbB]);
        o[n].v = WMMA_BF16(pf.v, vf, o[n].v);
      }
    }
    __syncthreads();   // all waves done with tile t before its buffer is re-staged
  }

  // normalize; write concat output A-swizzled over [BS, 1024]
#pragma unroll
  for (int r = 0; r < 8; ++r) {
    const float inv = 1.0f / l_run[r];
    const int g = b * SEQ + qbase + wave * 16 + mb + r;
#pragma unroll
    for (int n = 0; n < 4; ++n) {
      const int k = h * D_HEAD + n * 16 + colN;
      Csw[aswz(D_MODEL / 32, g, k)] = (bf16_t)(o[n].e[r] * inv);
    }
  }
}

// ---------------------------------------------------------------------------
// Output projection, LDS-free, software-pipelined: 32x64 tile per wave.
// ---------------------------------------------------------------------------
__global__ __launch_bounds__(128) void k_out(
    const bf16_t* __restrict__ Asw, const bf16_t* __restrict__ Wsw,
    float* __restrict__ Out) {
  const int rowBlk = blockIdx.x;        // 128 rows per block
  const int nbBase = blockIdx.y * 4;
  const int tid  = threadIdx.x;
  const int wave = tid >> 5;
  const int lane = tid & 31;
  const int rb0  = rowBlk * 8 + wave * 2;

  FragF32 acc[8] = {};
  gemm_tile_32x64<D_MODEL / 32>(Asw, Wsw, rb0, rb0 + 1, nbBase, lane, acc);

  const int colN = lane & 15;
  const int mb   = (lane >> 4) * 8;
#pragma unroll
  for (int half = 0; half < 2; ++half)
#pragma unroll
    for (int n = 0; n < 4; ++n)
#pragma unroll
      for (int r = 0; r < 8; ++r)
        Out[(size_t)(rowBlk * 128 + wave * 32 + half * 16 + mb + r) * D_MODEL +
            (nbBase + n) * 16 + colN] = acc[half * 4 + n].e[r];
}

// ---------------------------------------------------------------------------
// host-side launcher
// ---------------------------------------------------------------------------
extern "C" void kernel_launch(void* const* d_in, const int* in_sizes, int n_in,
                              void* d_out, int out_size, void* d_ws, size_t ws_size,
                              hipStream_t stream) {
  (void)in_sizes; (void)n_in; (void)out_size; (void)ws_size;

  const float* rs = (const float*)d_in[0];   // [B,S,M]
  const float* wq = (const float*)d_in[1];   // [H,M,D]
  const float* wk = (const float*)d_in[2];
  const float* wv = (const float*)d_in[3];
  const float* wo = (const float*)d_in[4];   // [M,M]
  float* out = (float*)d_out;

  bf16_t* p = (bf16_t*)d_ws;
  bf16_t* Xsw = p; p += (size_t)BS * D_MODEL;
  bf16_t* Wqs = p; p += (size_t)N_HEADS * D_MODEL * D_HEAD;
  bf16_t* Wks = p; p += (size_t)N_HEADS * D_MODEL * D_HEAD;
  bf16_t* Wvs = p; p += (size_t)N_HEADS * D_MODEL * D_HEAD;
  bf16_t* Wos = p; p += (size_t)D_MODEL * D_MODEL;
  bf16_t* Qb  = p; p += (size_t)BS * D_MODEL;
  bf16_t* Kb  = p; p += (size_t)BS * D_MODEL;
  bf16_t* Vb  = p; p += (size_t)BS * D_MODEL;
  bf16_t* Cb  = p; p += (size_t)BS * D_MODEL;

  k_cvt_swA<<<2048, 256, 0, stream>>>(rs, Xsw, BS, D_MODEL);
  k_cvt_swB<<<1024, 256, 0, stream>>>(wq, Wqs, N_HEADS, D_MODEL, D_HEAD);
  k_cvt_swB<<<1024, 256, 0, stream>>>(wk, Wks, N_HEADS, D_MODEL, D_HEAD);
  k_cvt_swB<<<1024, 256, 0, stream>>>(wv, Wvs, N_HEADS, D_MODEL, D_HEAD);
  k_cvt_swB<<<1024, 256, 0, stream>>>(wo, Wos, 1, D_MODEL, D_MODEL);

  k_qkv<<<dim3(BS / 128, N_HEADS, 3), 128, 0, stream>>>(Xsw, Wqs, Wks, Wvs, Qb, Kb, Vb);
  k_attn<<<dim3(SEQ / 64, N_HEADS, BATCH), 128, 0, stream>>>(Qb, Kb, Vb, Cb);
  k_out<<<dim3(BS / 128, D_MODEL / 64), 128, 0, stream>>>(Cb, Wos, out);
}
